// CrossAttention_52458730553754
// MI455X (gfx1250) — matrix operand order repaired and checked by reading, not compiled, not run on hardware
//
#include <hip/hip_runtime.h>
#include <hip/hip_bf16.h>

typedef __attribute__((ext_vector_type(16))) _Float16 v16h;
typedef __attribute__((ext_vector_type(8)))  _Float16 v8h;
typedef __attribute__((ext_vector_type(8)))  float    v8f;

#define B_   4
#define LQ_  1024
#define LK_  2048
#define DQ_  1024
#define DK_  512
#define DV_  512
#define D_   1024
#define H_   16
#define HD_  64

// ---------------------------------------------------------------------------
// WMMA fragment loaders (wave32; contiguous k order K = 16*(lane/16) + e, same for A and B)
// ---------------------------------------------------------------------------

// A-matrix 16x32 f16, row-major source with leading dim `lda` (in elements).
// lane -> M = lane&15 ; kb = (lane<16)?0:16
// vector elem e: K=kb+e
__device__ __forceinline__ v16h load_frag_a16x32(const _Float16* A, int lda, int lane) {
  const int m  = lane & 15;
  const int kb = (lane & 16) ? 16 : 0;
  const _Float16* p = A + (size_t)m * lda + kb;
  v8h lo = *reinterpret_cast<const v8h*>(p);
  v8h hi = *reinterpret_cast<const v8h*>(p + 8);
  v16h r;
#pragma unroll
  for (int i = 0; i < 8; ++i) { r[i] = lo[i]; r[i + 8] = hi[i]; }
  return r;
}

// B-matrix 32x16 f16 loaded from "BT" storage: BT[n][k] contiguous in k,
// leading dim `ldb`. lane -> N = lane&15 ; kb = (lane<16)?0:16 ; elem e -> K=kb+e
__device__ __forceinline__ v16h load_frag_b_nk(const _Float16* BT, int ldb, int lane) {
  const int n  = lane & 15;
  const int kb = (lane & 16) ? 16 : 0;
  const _Float16* p = BT + (size_t)n * ldb + kb;
  v8h lo = *reinterpret_cast<const v8h*>(p);
  v8h hi = *reinterpret_cast<const v8h*>(p + 8);
  v16h r;
#pragma unroll
  for (int i = 0; i < 8; ++i) { r[i] = lo[i]; r[i + 8] = hi[i]; }
  return r;
}

// Fragment bundle for a 32x64 C tile at one 32-wide K step
struct Frags {
  v16h a0, a1;   // rows m0..m0+15, m0+16..m0+31
  v16h b[4];     // 4 x 16-col subtiles
};

__device__ __forceinline__ Frags load_frags(const _Float16* A, const _Float16* BT,
                                            int K, int m0, int n0, int k0, int lane) {
  Frags f;
  f.a0 = load_frag_a16x32(A + (size_t)m0 * K + k0, K, lane);
  f.a1 = load_frag_a16x32(A + (size_t)(m0 + 16) * K + k0, K, lane);
#pragma unroll
  for (int t = 0; t < 4; ++t)
    f.b[t] = load_frag_b_nk(BT + (size_t)(n0 + t * 16) * K + k0, K, lane);
  return f;
}

// ---------------------------------------------------------------------------
// Prep kernels
// ---------------------------------------------------------------------------

__global__ __launch_bounds__(256) void layernorm_h(
    const float* __restrict__ x, const float* __restrict__ g,
    const float* __restrict__ bb, _Float16* __restrict__ y, int Dm) {
  __shared__ float red[256];
  const int row = blockIdx.x;
  const float* xr = x + (size_t)row * Dm;
  float s = 0.f;
  for (int i = threadIdx.x; i < Dm; i += 256) s += xr[i];
  red[threadIdx.x] = s; __syncthreads();
  for (int o = 128; o > 0; o >>= 1) {
    if (threadIdx.x < o) red[threadIdx.x] += red[threadIdx.x + o];
    __syncthreads();
  }
  const float mu = red[0] / (float)Dm;
  __syncthreads();
  float v = 0.f;
  for (int i = threadIdx.x; i < Dm; i += 256) { float d = xr[i] - mu; v += d * d; }
  red[threadIdx.x] = v; __syncthreads();
  for (int o = 128; o > 0; o >>= 1) {
    if (threadIdx.x < o) red[threadIdx.x] += red[threadIdx.x + o];
    __syncthreads();
  }
  const float rs = rsqrtf(red[0] / (float)Dm + 1e-5f);
  for (int i = threadIdx.x; i < Dm; i += 256)
    y[(size_t)row * Dm + i] = (_Float16)((xr[i] - mu) * rs * g[i] + bb[i]);
}

__global__ __launch_bounds__(256) void cvt_h(const float* __restrict__ x,
                                             _Float16* __restrict__ y, int n) {
  int i = blockIdx.x * 256 + threadIdx.x;
  if (i < n) y[i] = (_Float16)x[i];
}

// WT[n][k] = (f16) W[k][n] ; W is [Kd, Nd] row-major
__global__ __launch_bounds__(256) void transpose_cvt(const float* __restrict__ W,
                                                     _Float16* __restrict__ WT,
                                                     int Kd, int Nd) {
  int idx = blockIdx.x * 256 + threadIdx.x;
  if (idx < Kd * Nd) {
    int k = idx / Nd, n = idx % Nd;
    WT[(size_t)n * Kd + k] = (_Float16)W[idx];
  }
}

// ---------------------------------------------------------------------------
// WMMA GEMM: one wave -> 32x64 tile of C, 2-deep software pipeline.
// K must be a multiple of 64 (512 / 1024 here).
// ---------------------------------------------------------------------------

__global__ __launch_bounds__(32) void gemm_bias_h(
    const _Float16* __restrict__ A, const _Float16* __restrict__ BT,
    const float* __restrict__ bias, _Float16* __restrict__ C,
    int M, int N, int K) {
  const int lane = threadIdx.x;
  const int n0 = blockIdx.x * 64;
  const int m0 = blockIdx.y * 32;
  v8f acc[2][4] = {};

  auto mma8 = [&](const Frags& f) {
#pragma unroll
    for (int t = 0; t < 4; ++t) {
      acc[0][t] = __builtin_amdgcn_wmma_f32_16x16x32_f16(false, f.a0, false, f.b[t],
                                                         (short)0, acc[0][t], false, false);
      acc[1][t] = __builtin_amdgcn_wmma_f32_16x16x32_f16(false, f.a1, false, f.b[t],
                                                         (short)0, acc[1][t], false, false);
    }
  };

  Frags f0 = load_frags(A, BT, K, m0, n0, 0, lane);
  int k0 = 0;
  for (; k0 < K - 64; k0 += 64) {
    Frags f1 = load_frags(A, BT, K, m0, n0, k0 + 32, lane);
    mma8(f0);
    f0 = load_frags(A, BT, K, m0, n0, k0 + 64, lane);
    mma8(f1);
  }
  {
    Frags f1 = load_frags(A, BT, K, m0, n0, k0 + 32, lane);
    mma8(f0);
    mma8(f1);
  }

  const int nlo = lane & 15;
  const int mh  = (lane & 16) ? 8 : 0;
#pragma unroll
  for (int w = 0; w < 2; ++w)
#pragma unroll
    for (int t = 0; t < 4; ++t) {
      const int n = n0 + t * 16 + nlo;
      const float bv = bias[n];
#pragma unroll
      for (int r = 0; r < 8; ++r)
        C[(size_t)(m0 + w * 16 + mh + r) * N + n] = (_Float16)(acc[w][t][r] + bv);
    }
}

// Same GEMM with bias + fp32 residual epilogue, fp32 output
__global__ __launch_bounds__(32) void gemm_bias_res_f(
    const _Float16* __restrict__ A, const _Float16* __restrict__ BT,
    const float* __restrict__ bias, const float* __restrict__ res,
    float* __restrict__ out, int M, int N, int K) {
  const int lane = threadIdx.x;
  const int n0 = blockIdx.x * 64;
  const int m0 = blockIdx.y * 32;
  v8f acc[2][4] = {};

  auto mma8 = [&](const Frags& f) {
#pragma unroll
    for (int t = 0; t < 4; ++t) {
      acc[0][t] = __builtin_amdgcn_wmma_f32_16x16x32_f16(false, f.a0, false, f.b[t],
                                                         (short)0, acc[0][t], false, false);
      acc[1][t] = __builtin_amdgcn_wmma_f32_16x16x32_f16(false, f.a1, false, f.b[t],
                                                         (short)0, acc[1][t], false, false);
    }
  };

  Frags f0 = load_frags(A, BT, K, m0, n0, 0, lane);
  int k0 = 0;
  for (; k0 < K - 64; k0 += 64) {
    Frags f1 = load_frags(A, BT, K, m0, n0, k0 + 32, lane);
    mma8(f0);
    f0 = load_frags(A, BT, K, m0, n0, k0 + 64, lane);
    mma8(f1);
  }
  {
    Frags f1 = load_frags(A, BT, K, m0, n0, k0 + 32, lane);
    mma8(f0);
    mma8(f1);
  }

  const int nlo = lane & 15;
  const int mh  = (lane & 16) ? 8 : 0;
#pragma unroll
  for (int w = 0; w < 2; ++w)
#pragma unroll
    for (int t = 0; t < 4; ++t) {
      const int n = n0 + t * 16 + nlo;
      const float bv = bias[n];
#pragma unroll
      for (int r = 0; r < 8; ++r) {
        size_t idx = (size_t)(m0 + w * 16 + mh + r) * N + n;
        out[idx] = acc[w][t][r] + bv + res[idx];
      }
    }
}

// ---------------------------------------------------------------------------
// Flash attention: one wave per (b, h, 16-row q-tile); K-chunk = 64
// ---------------------------------------------------------------------------

__global__ __launch_bounds__(32) void attention_fwd(
    const _Float16* __restrict__ Qb, const _Float16* __restrict__ Kbf,
    const _Float16* __restrict__ Vbf, const int* __restrict__ mask,
    _Float16* __restrict__ ctx) {
  const int lane = threadIdx.x;
  const int q0 = blockIdx.x * 16;
  const int h  = blockIdx.y;
  const int b  = blockIdx.z;

  __shared__ __align__(16) float    S[16][64];
  __shared__ __align__(16) _Float16 P[16][64];
  __shared__ __align__(16) _Float16 Vs[64][72];   // padded to dodge bank conflicts
  __shared__ float mAdd[64];
  __shared__ float partMax[2][16], partSum[2][16];
  __shared__ float rowScale[16], rowM[16], rowL[16];

  const _Float16* Qbase = Qb + ((size_t)(b * LQ_ + q0)) * D_ + h * HD_;
  const v16h aq0 = load_frag_a16x32(Qbase, D_, lane);        // d 0..31
  const v16h aq1 = load_frag_a16x32(Qbase + 32, D_, lane);   // d 32..63

  if (lane < 16) { rowM[lane] = -1e30f; rowL[lane] = 0.f; }
  v8f o[4] = {};
  __syncthreads();

  const float scale = 0.125f;  // HD^-0.5
  const int row = lane & 15;
  const int ch  = lane >> 4;       // which 32-col half of the chunk this lane owns
  const int jlo = ch * 32;

  for (int kc = 0; kc < LK_; kc += 64) {
    // stage V chunk [64 x 64] f16 into LDS (coalesced 16B segments)
    {
      const _Float16* Vg = Vbf + ((size_t)(b * LK_ + kc)) * D_ + h * HD_;
      for (int i = lane; i < 64 * 8; i += 32) {
        int r = i >> 3, seg = i & 7;
        *reinterpret_cast<v8h*>(&Vs[r][seg * 8]) =
            *reinterpret_cast<const v8h*>(Vg + (size_t)r * D_ + seg * 8);
      }
      // key-padding mask -> additive vector
      for (int j = lane; j < 64; j += 32)
        mAdd[j] = (mask[b * LK_ + kc + j] == 0) ? -1e30f : 0.f;
    }
    // scores S = Q K^T for 4 k-subtiles of 16 (two WMMAs each over HD=64)
#pragma unroll
    for (int kt = 0; kt < 4; ++kt) {
      const _Float16* Kbase = Kbf + ((size_t)(b * LK_ + kc + kt * 16)) * D_ + h * HD_;
      v16h b0 = load_frag_b_nk(Kbase, D_, lane);
      v16h b1 = load_frag_b_nk(Kbase + 32, D_, lane);
      v8f c = {};
      c = __builtin_amdgcn_wmma_f32_16x16x32_f16(false, aq0, false, b0, (short)0, c, false, false);
      c = __builtin_amdgcn_wmma_f32_16x16x32_f16(false, aq1, false, b1, (short)0, c, false, false);
      const int n  = kt * 16 + (lane & 15);
      const int mh = (lane & 16) ? 8 : 0;
#pragma unroll
      for (int r = 0; r < 8; ++r) S[mh + r][n] = c[r];
    }
    __syncthreads();

    // online softmax, all 32 lanes: lane owns (row, 32-col half)
    float pmax = -1e30f;
    for (int j = jlo; j < jlo + 32; ++j) {
      float s = S[row][j] * scale + mAdd[j];
      S[row][j] = s;
      pmax = fmaxf(pmax, s);
    }
    partMax[ch][row] = pmax;
    __syncthreads();

    const float mPrev = rowM[row];
    const float mNew  = fmaxf(mPrev, fmaxf(partMax[0][row], partMax[1][row]));
    float psum = 0.f;
    for (int j = jlo; j < jlo + 32; ++j) {
      float p = __expf(S[row][j] - mNew);
      P[row][j] = (_Float16)p;
      psum += p;
    }
    partSum[ch][row] = psum;
    __syncthreads();

    if (lane < 16) {
      const float corr = __expf(mPrev - mNew);
      rowM[row] = mNew;
      rowL[row] = rowL[row] * corr + partSum[0][row] + partSum[1][row];
      rowScale[row] = corr;
    }
    __syncthreads();

    // rescale running output accumulators
    {
      const int mh = (lane & 16) ? 8 : 0;
      float fac[8];
#pragma unroll
      for (int r = 0; r < 8; ++r) fac[r] = rowScale[mh + r];
#pragma unroll
      for (int t = 0; t < 4; ++t)
#pragma unroll
        for (int r = 0; r < 8; ++r) o[t][r] *= fac[r];
    }

    // O += P @ V  (two K=32 halves, 4 d-subtiles)
#pragma unroll
    for (int hk = 0; hk < 2; ++hk) {
      v16h ap = load_frag_a16x32(&P[0][0] + hk * 32, 64, lane);
#pragma unroll
      for (int t = 0; t < 4; ++t) {
        const int n  = t * 16 + (lane & 15);
        const int kb = (lane & 16) ? 16 : 0;
        v16h bv;
#pragma unroll
        for (int e = 0; e < 16; ++e) bv[e] = Vs[hk * 32 + kb + e][n];
        o[t] = __builtin_amdgcn_wmma_f32_16x16x32_f16(false, ap, false, bv,
                                                      (short)0, o[t], false, false);
      }
    }
    __syncthreads();
  }

  if (lane < 16) rowScale[lane] = 1.f / rowL[lane];
  __syncthreads();
  {
    const int mh = (lane & 16) ? 8 : 0;
    float inv[8];
#pragma unroll
    for (int r = 0; r < 8; ++r) inv[r] = rowScale[mh + r];
    _Float16* cb = ctx + ((size_t)(b * LQ_ + q0)) * D_ + h * HD_;
#pragma unroll
    for (int t = 0; t < 4; ++t)
#pragma unroll
      for (int r = 0; r < 8; ++r)
        cb[(size_t)(mh + r) * D_ + t * 16 + (lane & 15)] = (_Float16)(o[t][r] * inv[r]);
  }
}

// ---------------------------------------------------------------------------
// Launch
// ---------------------------------------------------------------------------

extern "C" void kernel_launch(void* const* d_in, const int* in_sizes, int n_in,
                              void* d_out, int out_size, void* d_ws, size_t ws_size,
                              hipStream_t stream) {
  const float* query = (const float*)d_in[0];
  const float* key   = (const float*)d_in[1];
  const float* value = (const float*)d_in[2];
  const int*   mask  = (const int*)d_in[3];
  const float* Wq = (const float*)d_in[4];
  const float* bq = (const float*)d_in[5];
  const float* Wk = (const float*)d_in[6];
  const float* bk = (const float*)d_in[7];
  const float* Wv = (const float*)d_in[8];
  const float* bv = (const float*)d_in[9];
  const float* Wo = (const float*)d_in[10];
  const float* bo = (const float*)d_in[11];
  const float* ln_g = (const float*)d_in[12];
  const float* ln_b = (const float*)d_in[13];
  float* out = (float*)d_out;

  char* ws = (char*)d_ws;
  size_t off = 0;
  auto alloc = [&](size_t bytes) { char* p = ws + off; off += (bytes + 255) & ~(size_t)255; return p; };

  _Float16* qn   = (_Float16*)alloc((size_t)B_ * LQ_ * DQ_ * 2);
  _Float16* keyh = (_Float16*)alloc((size_t)B_ * LK_ * DK_ * 2);
  _Float16* valh = (_Float16*)alloc((size_t)B_ * LK_ * DV_ * 2);
  _Float16* WqT  = (_Float16*)alloc((size_t)D_ * DQ_ * 2);
  _Float16* WkT  = (_Float16*)alloc((size_t)D_ * DK_ * 2);
  _Float16* WvT  = (_Float16*)alloc((size_t)D_ * DV_ * 2);
  _Float16* WoT  = (_Float16*)alloc((size_t)D_ * D_ * 2);
  _Float16* Qb   = (_Float16*)alloc((size_t)B_ * LQ_ * D_ * 2);
  _Float16* Kb   = (_Float16*)alloc((size_t)B_ * LK_ * D_ * 2);
  _Float16* Vb   = (_Float16*)alloc((size_t)B_ * LK_ * D_ * 2);
  _Float16* Ctx  = (_Float16*)alloc((size_t)B_ * LQ_ * D_ * 2);

  // 1. LayerNorm(query) -> f16
  layernorm_h<<<B_ * LQ_, 256, 0, stream>>>(query, ln_g, ln_b, qn, DQ_);

  // 2. key/value -> f16
  {
    int n = B_ * LK_ * DK_;
    cvt_h<<<(n + 255) / 256, 256, 0, stream>>>(key, keyh, n);
    n = B_ * LK_ * DV_;
    cvt_h<<<(n + 255) / 256, 256, 0, stream>>>(value, valh, n);
  }

  // 3. weights -> transposed f16 ([N][K])
  transpose_cvt<<<(DQ_ * D_ + 255) / 256, 256, 0, stream>>>(Wq, WqT, DQ_, D_);
  transpose_cvt<<<(DK_ * D_ + 255) / 256, 256, 0, stream>>>(Wk, WkT, DK_, D_);
  transpose_cvt<<<(DV_ * D_ + 255) / 256, 256, 0, stream>>>(Wv, WvT, DV_, D_);
  transpose_cvt<<<(D_ * D_  + 255) / 256, 256, 0, stream>>>(Wo, WoT, D_, D_);

  // 4. projections (WMMA GEMMs, 32x64 tiles)
  gemm_bias_h<<<dim3(D_ / 64, (B_ * LQ_) / 32), 32, 0, stream>>>(qn,   WqT, bq, Qb, B_ * LQ_, D_, DQ_);
  gemm_bias_h<<<dim3(D_ / 64, (B_ * LK_) / 32), 32, 0, stream>>>(keyh, WkT, bk, Kb, B_ * LK_, D_, DK_);
  gemm_bias_h<<<dim3(D_ / 64, (B_ * LK_) / 32), 32, 0, stream>>>(valh, WvT, bv, Vb, B_ * LK_, D_, DV_);

  // 5. attention (WMMA flash)
  attention_fwd<<<dim3(LQ_ / 16, H_, B_), 32, 0, stream>>>(Qb, Kb, Vb, mask, Ctx);

  // 6. output projection + bias + residual (fp32 out)
  gemm_bias_res_f<<<dim3(D_ / 64, (B_ * LQ_) / 32), 32, 0, stream>>>(Ctx, WoT, bo, query, out, B_ * LQ_, D_, D_);
}
